// PointNextEncoder_50500225466725
// MI455X (gfx1250) — matrix operand order repaired and checked by reading, not compiled, not run on hardware
//
#include <hip/hip_runtime.h>
#include <hip/hip_bf16.h>
#include <math.h>

typedef _Float16 f16;
typedef __attribute__((ext_vector_type(16))) _Float16 v16h;
typedef __attribute__((ext_vector_type(8)))  _Float16 v8h;
typedef __attribute__((ext_vector_type(8)))  float    v8f;

union H16 { v16h v; v8h h2[2]; };

#define NB 8   // batch

__device__ __forceinline__ float geluf(float x) {
  return 0.5f * x * (1.0f + erff(x * 0.70710678118654752f));
}

// ---------------------------------------------------------------------------
// WMMA GEMM:  C[rows x cols] = A[rows x kpad](f16) * Bt[cols x kpad](f16)^T + bias
// One wave computes a 16x64 strip: 4 WMMA tiles sharing one A fragment.
// A fragment (16x32 f16): lane m=lane&15, kh=lane>>4; halves [k0+kh*8..+7] and
// [k0+16+kh*8..+7] (per ISA 16-bit A layout).  B fragment (32x16): lane n=lane&15,
// kh=lane>>4; halves [k0+kh*8..+7] and [k0+16+kh*8..+7] (same as A).
// ---------------------------------------------------------------------------
__global__ __launch_bounds__(256)
void gemm_wmma(const f16* __restrict__ A, const f16* __restrict__ Bt,
               const float* __restrict__ bias, float* __restrict__ C,
               int rows, int cols, int kpad) {
  int wid  = (int)(((long)blockIdx.x * blockDim.x + threadIdx.x) >> 5);
  int lane = threadIdx.x & 31;
  int tilesM  = rows >> 4;
  int tilesN4 = cols >> 6;
  if (wid >= tilesM * tilesN4) return;
  int tm  = wid / tilesN4;
  int tn4 = wid - tm * tilesN4;
  int m  = lane & 15;
  int kh = lane >> 4;
  const f16* aRow = A  + (size_t)(tm * 16 + m) * kpad;
  const f16* bRow = Bt + (size_t)(tn4 * 64 + m) * kpad;
  v8f acc0 = {}, acc1 = {}, acc2 = {}, acc3 = {};
  for (int k0 = 0; k0 < kpad; k0 += 32) {
    H16 a, b0, b1, b2, b3;
    a.h2[0] = *(const v8h*)(aRow + k0 + kh * 8);
    a.h2[1] = *(const v8h*)(aRow + k0 + 16 + kh * 8);
    const f16* bp = bRow + k0 + kh * 8;
    b0.h2[0] = *(const v8h*)(bp);
    b0.h2[1] = *(const v8h*)(bp + 16);
    b1.h2[0] = *(const v8h*)(bp + (size_t)16 * kpad);
    b1.h2[1] = *(const v8h*)(bp + (size_t)16 * kpad + 16);
    b2.h2[0] = *(const v8h*)(bp + (size_t)32 * kpad);
    b2.h2[1] = *(const v8h*)(bp + (size_t)32 * kpad + 16);
    b3.h2[0] = *(const v8h*)(bp + (size_t)48 * kpad);
    b3.h2[1] = *(const v8h*)(bp + (size_t)48 * kpad + 16);
    acc0 = __builtin_amdgcn_wmma_f32_16x16x32_f16(false, a.v, false, b0.v, (short)0, acc0, false, false);
    acc1 = __builtin_amdgcn_wmma_f32_16x16x32_f16(false, a.v, false, b1.v, (short)0, acc1, false, false);
    acc2 = __builtin_amdgcn_wmma_f32_16x16x32_f16(false, a.v, false, b2.v, (short)0, acc2, false, false);
    acc3 = __builtin_amdgcn_wmma_f32_16x16x32_f16(false, a.v, false, b3.v, (short)0, acc3, false, false);
  }
  int colBase = tn4 * 64 + m;
  float bb0 = bias[colBase], bb1 = bias[colBase + 16];
  float bb2 = bias[colBase + 32], bb3 = bias[colBase + 48];
#pragma unroll
  for (int r = 0; r < 8; ++r) {
    int row = tm * 16 + kh * 8 + r;   // C/D layout: VGPR r -> M = r + 8*(lane>>4)
    float* cp = C + (size_t)row * cols + colBase;
    cp[0]  = acc0[r] + bb0;
    cp[16] = acc1[r] + bb1;
    cp[32] = acc2[r] + bb2;
    cp[48] = acc3[r] + bb3;
  }
}

// ---------------------------------------------------------------------------
// Rowwise LayerNorm (+optional exact GeLU). One wave per row (cols multiple of 32,
// <= 512). Optionally emits f32 (may alias X) and/or f16 padded to kpadOut.
// ---------------------------------------------------------------------------
__global__ void ln_act_kern(const float* X, const float* __restrict__ gg,
                            const float* __restrict__ be, float* outF,
                            f16* outH, int rows, int cols, int kpadOut, int act) {
  int row  = (int)(((long)blockIdx.x * blockDim.x + threadIdx.x) >> 5);
  int lane = threadIdx.x & 31;
  if (row >= rows) return;
  const float* x = X + (size_t)row * cols;
  int nPer = cols >> 5;
  float v[16];
  float s = 0.f;
  for (int i = 0; i < nPer; ++i) { v[i] = x[lane + (i << 5)]; s += v[i]; }
  for (int off = 16; off; off >>= 1) s += __shfl_xor(s, off, 32);
  float mu = s / (float)cols;
  float q = 0.f;
  for (int i = 0; i < nPer; ++i) { float c = v[i] - mu; q += c * c; }
  for (int off = 16; off; off >>= 1) q += __shfl_xor(q, off, 32);
  float rs = rsqrtf(q / (float)cols + 1e-5f);
  for (int i = 0; i < nPer; ++i) {
    int c = lane + (i << 5);
    float y = (v[i] - mu) * rs * gg[c] + be[c];
    if (act) y = geluf(y);
    if (outF) outF[(size_t)row * cols + c] = y;
    if (outH) outH[(size_t)row * kpadOut + c] = (f16)y;
  }
  if (outH)
    for (int c = cols + lane; c < kpadOut; c += 32)
      outH[(size_t)row * kpadOut + c] = (f16)0.0f;
}

// ---------------------------------------------------------------------------
// Farthest point sampling: one workgroup (512 thr) per batch; dist in LDS,
// serial m-step loop with wave32 shuffle argmax + cross-wave LDS reduce.
// ---------------------------------------------------------------------------
__global__ void fps_kernel(const float* __restrict__ xyz, int* __restrict__ fidx,
                           int N, int M) {
  __shared__ float dist[4096];
  __shared__ float bv_s[16];
  __shared__ int   bi_s[16];
  __shared__ int   lastIdx;
  int b = blockIdx.x;
  const float* P = xyz + (long)b * N * 3;
  for (int i = threadIdx.x; i < N; i += blockDim.x) dist[i] = 1e10f;
  if (threadIdx.x == 0) lastIdx = 0;
  __syncthreads();
  int nw = blockDim.x >> 5;
  for (int it = 0; it < M; ++it) {
    int last = lastIdx;
    if (threadIdx.x == 0) fidx[(long)b * M + it] = last;
    float lx = P[last * 3], ly = P[last * 3 + 1], lz = P[last * 3 + 2];
    float bv = -1.0f; int bi = 0x7fffffff;
    for (int i = threadIdx.x; i < N; i += blockDim.x) {
      float dx = P[i * 3] - lx, dy = P[i * 3 + 1] - ly, dz = P[i * 3 + 2] - lz;
      float d = fminf(dist[i], dx * dx + dy * dy + dz * dz);
      dist[i] = d;
      if (d > bv || (d == bv && i < bi)) { bv = d; bi = i; }
    }
    for (int off = 16; off; off >>= 1) {
      float ov = __shfl_xor(bv, off, 32);
      int   oi = __shfl_xor(bi, off, 32);
      if (ov > bv || (ov == bv && oi < bi)) { bv = ov; bi = oi; }
    }
    __syncthreads();
    if ((threadIdx.x & 31) == 0) { bv_s[threadIdx.x >> 5] = bv; bi_s[threadIdx.x >> 5] = bi; }
    __syncthreads();
    if (threadIdx.x == 0) {
      float v = bv_s[0]; int ix = bi_s[0];
      for (int w = 1; w < nw; ++w)
        if (bv_s[w] > v || (bv_s[w] == v && bi_s[w] < ix)) { v = bv_s[w]; ix = bi_s[w]; }
      lastIdx = ix;
    }
    __syncthreads();
  }
}

// Ball query == first K passing indices in ascending order (matches sort-based ref)
__global__ void ball_query_k(const float* __restrict__ xyz, const float* __restrict__ ctr,
                             int* __restrict__ nidx, long BM, int N, int M, int K, float r2) {
  long t = (long)blockIdx.x * blockDim.x + threadIdx.x;
  if (t >= BM) return;
  int b = (int)(t / M);
  const float* P = xyz + (long)b * N * 3;
  float cx = ctr[t * 3], cy = ctr[t * 3 + 1], cz = ctr[t * 3 + 2];
  int* o = nidx + t * K;
  int cnt = 0, first = 0; bool got = false;
  for (int j = 0; j < N; ++j) {
    float dx = P[j * 3] - cx, dy = P[j * 3 + 1] - cy, dz = P[j * 3 + 2] - cz;
    if (dx * dx + dy * dy + dz * dz <= r2) {
      if (!got) { got = true; first = j; }
      o[cnt++] = j;
      if (cnt == K) break;
    }
  }
  for (; cnt < K; ++cnt) o[cnt] = first;
}

// Build grouped-input rows: [rel(3) | feat(C) | pef(16) | pad] as f16
__global__ void build_gi(const float* __restrict__ xyz, const float* __restrict__ ctr,
                         const int* __restrict__ nidx, const float* __restrict__ feat,
                         const float* __restrict__ pew, const float* __restrict__ peb,
                         float invR, f16* __restrict__ gi, long bm0, int cnt,
                         int M, int K, int N, int C, int kpad) {
  long t = (long)blockIdx.x * blockDim.x + threadIdx.x;
  long total = (long)cnt * K * kpad;
  if (t >= total) return;
  int col = (int)(t % kpad);
  long rl = t / kpad;
  int k = (int)(rl % K);
  long bm = bm0 + rl / K;
  int b = (int)(bm / M);
  float val = 0.0f;
  if (col < 3 + C + 16) {
    int j = nidx[bm * K + k];
    const float* pj = xyz + ((long)b * N + j) * 3;
    if (col >= 3 && col < 3 + C) {
      val = feat[((long)b * N + j) * C + (col - 3)];
    } else {
      const float* pcn = ctr + bm * 3;
      float rx = (pj[0] - pcn[0]) * invR;
      float ry = (pj[1] - pcn[1]) * invR;
      float rz = (pj[2] - pcn[2]) * invR;
      if (col < 3) val = (col == 0) ? rx : ((col == 1) ? ry : rz);
      else {
        int q = col - 3 - C;
        val = rx * pew[q] + ry * pew[16 + q] + rz * pew[32 + q] + peb[q];
      }
    }
  }
  gi[t] = (f16)val;
}

__global__ void cvt_weight(const float* __restrict__ W, f16* __restrict__ Wt,
                           int Cin, int Cout, int kpad) {
  long t = (long)blockIdx.x * blockDim.x + threadIdx.x;
  if (t >= (long)Cout * kpad) return;
  int k = (int)(t % kpad);
  int n = (int)(t / kpad);
  Wt[t] = (k < Cin) ? (f16)W[(long)k * Cout + n] : (f16)0.0f;
}

__global__ void strip_xyz(const float* __restrict__ pc, float* __restrict__ xyz, long npts) {
  long t = (long)blockIdx.x * blockDim.x + threadIdx.x;
  if (t >= npts * 3) return;
  long p = t / 3; int a = (int)(t % 3);
  xyz[t] = pc[p * 6 + a];
}

__global__ void build_stage0(const float* __restrict__ pc, f16* __restrict__ A,
                             long rows, int kpad) {
  long t = (long)blockIdx.x * blockDim.x + threadIdx.x;
  if (t >= rows * kpad) return;
  long r = t / kpad; int c = (int)(t % kpad);
  A[t] = (c < 6) ? (f16)pc[r * 6 + c] : (f16)0.0f;
}

__global__ void gather_xyz(const float* __restrict__ xyz, const int* __restrict__ fidx,
                           float* __restrict__ ctr, long BM, int M, int N) {
  long t = (long)blockIdx.x * blockDim.x + threadIdx.x;
  if (t >= BM * 3) return;
  long bm = t / 3; int a = (int)(t % 3);
  int b = (int)(bm / M);
  int j = fidx[bm];
  ctr[t] = xyz[((long)b * N + j) * 3 + a];
}

__global__ void gather_feat_h(const float* __restrict__ feat, const int* __restrict__ fidx,
                              f16* __restrict__ out, long BM, int M, int N, int C, int kpad) {
  long t = (long)blockIdx.x * blockDim.x + threadIdx.x;
  if (t >= BM * kpad) return;
  long bm = t / kpad; int c = (int)(t % kpad);
  int b = (int)(bm / M);
  int j = fidx[bm];
  out[t] = (c < C) ? (f16)feat[((long)b * N + j) * C + c] : (f16)0.0f;
}

__global__ void cvt_h(const float* __restrict__ x, f16* __restrict__ out,
                      long rows, int C, int kpad) {
  long t = (long)blockIdx.x * blockDim.x + threadIdx.x;
  if (t >= rows * kpad) return;
  long r = t / kpad; int c = (int)(t % kpad);
  out[t] = (c < C) ? (f16)x[r * C + c] : (f16)0.0f;
}

__global__ void max_k(const float* __restrict__ X, float* __restrict__ out,
                      long cnt, int K, int C) {
  long t = (long)blockIdx.x * blockDim.x + threadIdx.x;
  if (t >= cnt * C) return;
  long bm = t / C; int c = (int)(t % C);
  const float* p = X + (bm * K) * (long)C + c;
  float m = -3.402823466e38f;
  for (int k = 0; k < K; ++k) m = fmaxf(m, p[(long)k * C]);
  out[t] = m;
}

__global__ void add_gelu(const float* a, const float* b, float* o, long n) {
  long t = (long)blockIdx.x * blockDim.x + threadIdx.x;
  if (t >= n) return;
  o[t] = geluf(a[t] + b[t]);
}

__global__ void pool_max(const float* __restrict__ feat, float* __restrict__ out,
                         int Bn, int M, int C) {
  long t = (long)blockIdx.x * blockDim.x + threadIdx.x;
  if (t >= (long)Bn * C) return;
  int b = (int)(t / C), c = (int)(t % C);
  float m = -3.402823466e38f;
  for (int i = 0; i < M; ++i) m = fmaxf(m, feat[((long)b * M + i) * C + c]);
  out[t] = m;
}

__global__ void mean_xyz(const float* __restrict__ xyz, float* __restrict__ out,
                         int Bn, int M) {
  long t = (long)blockIdx.x * blockDim.x + threadIdx.x;
  if (t >= (long)Bn * 3) return;
  int b = (int)(t / 3), a = (int)(t % 3);
  float s = 0.f;
  for (int i = 0; i < M; ++i) s += xyz[((long)b * M + i) * 3 + a];
  out[t] = s / (float)M;
}

__global__ void sin_build(const float* __restrict__ px, f16* __restrict__ A,
                          int rows, int dim) {
  long t = (long)blockIdx.x * blockDim.x + threadIdx.x;
  if (t >= (long)rows * dim) return;
  int r = (int)(t / dim), c = (int)(t % dim);
  float v = 0.f;
  if (r < NB) {
    int axis = c / 32, j = c % 32;
    int jj = (j < 16) ? j : j - 16;
    float fr = expf(-9.210340371976184f * (float)jj / 15.0f);
    float a = px[r * 3 + axis] * fr;
    v = (j < 16) ? sinf(a) : cosf(a);
  }
  A[t] = (f16)v;
}

__global__ void add_cvt(const float* a, const float* b, f16* o, int rows, int C) {
  long t = (long)blockIdx.x * blockDim.x + threadIdx.x;
  if (t >= (long)rows * C) return;
  int r = (int)(t / C);
  o[t] = (r < NB) ? (f16)(a[t] + b[t]) : (f16)0.0f;
}

// ---------------------------------------------------------------------------
// Host side
// ---------------------------------------------------------------------------
struct PM { const float* b; const float* be; const float* g; const float* w; };
static PM getPM(void* const* d_in, int base) {
  return { (const float*)d_in[base], (const float*)d_in[base + 1],
           (const float*)d_in[base + 2], (const float*)d_in[base + 3] };
}

struct Bump {
  char* base; size_t used; size_t cap;
  void* get(size_t bytes) {
    size_t o = (used + 255) & ~(size_t)255;
    used = o + bytes;
    return base + o;
  }
};

struct Scratch { f16* gi; float* tA; f16* tH; float* tB; };

static inline unsigned gdiv(long t) { return (unsigned)((t + 255) / 256); }

static void grouped_mlp(hipStream_t s, const Scratch& sc,
                        const float* xyzSrc, const float* ctr, const int* nidx,
                        const float* feat, int Nsrc, int M, int K, int C,
                        const float* pew, const float* peb, float radius,
                        const f16* W1t, const PM& L1, int C1,
                        const f16* W2t, const PM& L2, int C2,
                        float* outBuf, int CH) {
  int kpadIn = (3 + C + 16 + 31) & ~31;
  for (long bm0 = 0; bm0 < (long)NB * M; bm0 += CH) {
    int cnt = CH;
    long rowsC = (long)cnt * K;
    build_gi<<<gdiv(rowsC * kpadIn), 256, 0, s>>>(xyzSrc, ctr, nidx, feat, pew, peb,
        1.0f / radius, sc.gi, bm0, cnt, M, K, Nsrc, C, kpadIn);
    long w1 = (rowsC / 16) * (C1 / 64);
    gemm_wmma<<<gdiv(w1 * 32), 256, 0, s>>>(sc.gi, W1t, L1.b, sc.tA, (int)rowsC, C1, kpadIn);
    ln_act_kern<<<gdiv(rowsC * 32), 256, 0, s>>>(sc.tA, L1.g, L1.be, nullptr, sc.tH,
                                                 (int)rowsC, C1, C1, 1);
    long w2 = (rowsC / 16) * (C2 / 64);
    gemm_wmma<<<gdiv(w2 * 32), 256, 0, s>>>(sc.tH, W2t, L2.b, sc.tB, (int)rowsC, C2, C1);
    ln_act_kern<<<gdiv(rowsC * 32), 256, 0, s>>>(sc.tB, L2.g, L2.be, sc.tB, nullptr,
                                                 (int)rowsC, C2, 0, 0);
    max_k<<<gdiv((long)cnt * C2), 256, 0, s>>>(sc.tB, outBuf + bm0 * C2, cnt, K, C2);
  }
}

extern "C" void kernel_launch(void* const* d_in, const int* in_sizes, int n_in,
                              void* d_out, int out_size, void* d_ws, size_t ws_size,
                              hipStream_t stream) {
  (void)in_sizes; (void)n_in; (void)out_size;
  const float* pc = (const float*)d_in[0];
  // params flattened in jax pytree order (dict keys sorted):
  PM gpp   = getPM(d_in, 1);
  PM outP  = getPM(d_in, 5);
  PM i1ch0 = getPM(d_in, 9);
  PM i1ch1 = getPM(d_in, 13);
  PM i1la0 = getPM(d_in, 17);
  PM i1la1 = getPM(d_in, 21);
  const float* i1peb = (const float*)d_in[25];
  const float* i1pew = (const float*)d_in[26];
  PM s1m0 = getPM(d_in, 27);
  PM s1m1 = getPM(d_in, 31);
  const float* s1peb = (const float*)d_in[35];
  const float* s1pew = (const float*)d_in[36];
  PM s1res = getPM(d_in, 37);
  PM i2ch0 = getPM(d_in, 41);
  PM i2ch1 = getPM(d_in, 45);
  PM i2la0 = getPM(d_in, 49);
  PM i2la1 = getPM(d_in, 53);
  const float* i2peb = (const float*)d_in[57];
  const float* i2pew = (const float*)d_in[58];
  PM s2m0 = getPM(d_in, 59);
  PM s2m1 = getPM(d_in, 63);
  const float* s2peb = (const float*)d_in[67];
  const float* s2pew = (const float*)d_in[68];
  PM s2res = getPM(d_in, 69);
  PM st0 = getPM(d_in, 73);

  Bump ws{ (char*)d_ws, 0, ws_size };

  // --- allocate weight buffers (conversions launched after size check) ---
  struct WCvt { const float* W; f16* out; int Cin, Cout, kpad; };
  WCvt wl[17]; int nwl = 0;
  auto allocW = [&](const float* W, int Cin, int Cout) -> f16* {
    int kpad = (Cin + 31) & ~31;
    f16* o = (f16*)ws.get((size_t)Cout * kpad * sizeof(f16));
    wl[nwl].W = W; wl[nwl].out = o; wl[nwl].Cin = Cin; wl[nwl].Cout = Cout; wl[nwl].kpad = kpad;
    ++nwl;
    return o;
  };
  f16* Wst0   = allocW(st0.w,   6, 64);
  f16* Ws1m0  = allocW(s1m0.w,  83, 64);
  f16* Ws1m1  = allocW(s1m1.w,  64, 128);
  f16* Ws1res = allocW(s1res.w, 64, 128);
  f16* Wi1la0 = allocW(i1la0.w, 147, 256);
  f16* Wi1la1 = allocW(i1la1.w, 256, 128);
  f16* Wi1ch0 = allocW(i1ch0.w, 128, 256);
  f16* Wi1ch1 = allocW(i1ch1.w, 256, 128);
  f16* Ws2m0  = allocW(s2m0.w,  147, 128);
  f16* Ws2m1  = allocW(s2m1.w,  128, 256);
  f16* Ws2res = allocW(s2res.w, 128, 256);
  f16* Wi2la0 = allocW(i2la0.w, 275, 512);
  f16* Wi2la1 = allocW(i2la1.w, 512, 256);
  f16* Wi2ch0 = allocW(i2ch0.w, 256, 512);
  f16* Wi2ch1 = allocW(i2ch1.w, 512, 256);
  f16* Wgpp   = allocW(gpp.w,   96, 256);
  f16* Wout   = allocW(outP.w,  256, 256);

  // --- persistent buffers ---
  float* xyz0   = (float*)ws.get((size_t)NB * 4096 * 3 * 4);
  float* feat0  = (float*)ws.get((size_t)32768 * 64 * 4);
  int*   fidx1  = (int*)ws.get((size_t)16384 * 4);
  float* xyz1   = (float*)ws.get((size_t)16384 * 3 * 4);
  int*   nidx1  = (int*)ws.get((size_t)16384 * 24 * 4);
  float* mo1    = (float*)ws.get((size_t)16384 * 128 * 4);
  float* feat1  = (float*)ws.get((size_t)16384 * 128 * 4);
  int*   nidx2  = (int*)ws.get((size_t)16384 * 24 * 4);
  float* agg1   = (float*)ws.get((size_t)16384 * 128 * 4);
  float* feat1b = (float*)ws.get((size_t)16384 * 128 * 4);
  int*   fidx2  = (int*)ws.get((size_t)8192 * 4);
  float* xyz2   = (float*)ws.get((size_t)8192 * 3 * 4);
  int*   nidx3  = (int*)ws.get((size_t)8192 * 32 * 4);
  float* mo2    = (float*)ws.get((size_t)8192 * 256 * 4);
  float* feat2  = (float*)ws.get((size_t)8192 * 256 * 4);
  int*   nidx4  = (int*)ws.get((size_t)8192 * 32 * 4);
  float* agg2   = (float*)ws.get((size_t)8192 * 256 * 4);
  float* feat2b = (float*)ws.get((size_t)8192 * 256 * 4);
  float* pooled = (float*)ws.get((size_t)NB * 256 * 4);
  float* pxyz   = (float*)ws.get((size_t)NB * 3 * 4);
  f16*   sinA   = (f16*)ws.get((size_t)16 * 96 * 2);
  float* gT     = (float*)ws.get((size_t)16 * 256 * 4);
  float* gBuf   = (float*)ws.get((size_t)NB * 256 * 4);
  f16*   A2h    = (f16*)ws.get((size_t)16 * 256 * 2);
  float* outT   = (float*)ws.get((size_t)16 * 256 * 4);
  // --- chunk scratch ---
  Scratch sc;
  sc.gi = (f16*)ws.get((size_t)16384 * 288 * 2);
  sc.tA = (float*)ws.get((size_t)16384 * 512 * 4);
  sc.tH = (f16*)ws.get((size_t)16384 * 512 * 2);
  sc.tB = (float*)ws.get((size_t)16384 * 256 * 4);

  if (ws.used > ws_size) return;  // insufficient workspace

  for (int i = 0; i < nwl; ++i) {
    long t = (long)wl[i].Cout * wl[i].kpad;
    cvt_weight<<<gdiv(t), 256, 0, stream>>>(wl[i].W, wl[i].out, wl[i].Cin, wl[i].Cout, wl[i].kpad);
  }

  // ---- stage0: pointwise MLP 6->64 ----
  strip_xyz<<<gdiv((long)32768 * 3), 256, 0, stream>>>(pc, xyz0, 32768L);
  build_stage0<<<gdiv((long)32768 * 32), 256, 0, stream>>>(pc, sc.gi, 32768L, 32);
  gemm_wmma<<<gdiv((long)(32768 / 16) * 1 * 32), 256, 0, stream>>>(sc.gi, Wst0, st0.b, sc.tA, 32768, 64, 32);
  ln_act_kern<<<gdiv((long)32768 * 32), 256, 0, stream>>>(sc.tA, st0.g, st0.be, feat0, nullptr, 32768, 64, 0, 1);

  // ---- s1_sa: 4096 -> 2048, r=0.08, k=24 ----
  fps_kernel<<<NB, 512, 0, stream>>>(xyz0, fidx1, 4096, 2048);
  gather_xyz<<<gdiv((long)16384 * 3), 256, 0, stream>>>(xyz0, fidx1, xyz1, 16384L, 2048, 4096);
  ball_query_k<<<gdiv(16384L), 256, 0, stream>>>(xyz0, xyz1, nidx1, 16384L, 4096, 2048, 24, 0.08f * 0.08f);
  grouped_mlp(stream, sc, xyz0, xyz1, nidx1, feat0, 4096, 2048, 24, 64,
              s1pew, s1peb, 0.08f, Ws1m0, s1m0, 64, Ws1m1, s1m1, 128, mo1, 1024);
  gather_feat_h<<<gdiv((long)16384 * 64), 256, 0, stream>>>(feat0, fidx1, sc.tH, 16384L, 2048, 4096, 64, 64);
  gemm_wmma<<<gdiv((long)(16384 / 16) * 2 * 32), 256, 0, stream>>>(sc.tH, Ws1res, s1res.b, sc.tA, 16384, 128, 64);
  ln_act_kern<<<gdiv((long)16384 * 32), 256, 0, stream>>>(sc.tA, s1res.g, s1res.be, sc.tA, nullptr, 16384, 128, 0, 0);
  add_gelu<<<gdiv((long)16384 * 128), 256, 0, stream>>>(mo1, sc.tA, feat1, (long)16384 * 128);

  // ---- s1_irb: 2048 pts, r=0.08, k=24 ----
  ball_query_k<<<gdiv(16384L), 256, 0, stream>>>(xyz1, xyz1, nidx2, 16384L, 2048, 2048, 24, 0.08f * 0.08f);
  grouped_mlp(stream, sc, xyz1, xyz1, nidx2, feat1, 2048, 2048, 24, 128,
              i1pew, i1peb, 0.08f, Wi1la0, i1la0, 256, Wi1la1, i1la1, 128, agg1, 1024);
  cvt_h<<<gdiv((long)16384 * 128), 256, 0, stream>>>(agg1, sc.tH, 16384L, 128, 128);
  gemm_wmma<<<gdiv((long)(16384 / 16) * 4 * 32), 256, 0, stream>>>(sc.tH, Wi1ch0, i1ch0.b, sc.tA, 16384, 256, 128);
  ln_act_kern<<<gdiv((long)16384 * 32), 256, 0, stream>>>(sc.tA, i1ch0.g, i1ch0.be, nullptr, sc.tH, 16384, 256, 256, 1);
  gemm_wmma<<<gdiv((long)(16384 / 16) * 2 * 32), 256, 0, stream>>>(sc.tH, Wi1ch1, i1ch1.b, sc.tB, 16384, 128, 256);
  ln_act_kern<<<gdiv((long)16384 * 32), 256, 0, stream>>>(sc.tB, i1ch1.g, i1ch1.be, sc.tB, nullptr, 16384, 128, 0, 0);
  add_gelu<<<gdiv((long)16384 * 128), 256, 0, stream>>>(sc.tB, feat1, feat1b, (long)16384 * 128);

  // ---- s2_sa: 2048 -> 1024, r=0.16, k=32 ----
  fps_kernel<<<NB, 512, 0, stream>>>(xyz1, fidx2, 2048, 1024);
  gather_xyz<<<gdiv((long)8192 * 3), 256, 0, stream>>>(xyz1, fidx2, xyz2, 8192L, 1024, 2048);
  ball_query_k<<<gdiv(8192L), 256, 0, stream>>>(xyz1, xyz2, nidx3, 8192L, 2048, 1024, 32, 0.16f * 0.16f);
  grouped_mlp(stream, sc, xyz1, xyz2, nidx3, feat1b, 2048, 1024, 32, 128,
              s2pew, s2peb, 0.16f, Ws2m0, s2m0, 128, Ws2m1, s2m1, 256, mo2, 512);
  gather_feat_h<<<gdiv((long)8192 * 128), 256, 0, stream>>>(feat1b, fidx2, sc.tH, 8192L, 1024, 2048, 128, 128);
  gemm_wmma<<<gdiv((long)(8192 / 16) * 4 * 32), 256, 0, stream>>>(sc.tH, Ws2res, s2res.b, sc.tA, 8192, 256, 128);
  ln_act_kern<<<gdiv((long)8192 * 32), 256, 0, stream>>>(sc.tA, s2res.g, s2res.be, sc.tA, nullptr, 8192, 256, 0, 0);
  add_gelu<<<gdiv((long)8192 * 256), 256, 0, stream>>>(mo2, sc.tA, feat2, (long)8192 * 256);

  // ---- s2_irb: 1024 pts, r=0.16, k=32 ----
  ball_query_k<<<gdiv(8192L), 256, 0, stream>>>(xyz2, xyz2, nidx4, 8192L, 1024, 1024, 32, 0.16f * 0.16f);
  grouped_mlp(stream, sc, xyz2, xyz2, nidx4, feat2, 1024, 1024, 32, 256,
              i2pew, i2peb, 0.16f, Wi2la0, i2la0, 512, Wi2la1, i2la1, 256, agg2, 512);
  cvt_h<<<gdiv((long)8192 * 256), 256, 0, stream>>>(agg2, sc.tH, 8192L, 256, 256);
  gemm_wmma<<<gdiv((long)(8192 / 16) * 8 * 32), 256, 0, stream>>>(sc.tH, Wi2ch0, i2ch0.b, sc.tA, 8192, 512, 256);
  ln_act_kern<<<gdiv((long)8192 * 32), 256, 0, stream>>>(sc.tA, i2ch0.g, i2ch0.be, nullptr, sc.tH, 8192, 512, 512, 1);
  gemm_wmma<<<gdiv((long)(8192 / 16) * 4 * 32), 256, 0, stream>>>(sc.tH, Wi2ch1, i2ch1.b, sc.tB, 8192, 256, 512);
  ln_act_kern<<<gdiv((long)8192 * 32), 256, 0, stream>>>(sc.tB, i2ch1.g, i2ch1.be, sc.tB, nullptr, 8192, 256, 0, 0);
  add_gelu<<<gdiv((long)8192 * 256), 256, 0, stream>>>(sc.tB, feat2, feat2b, (long)8192 * 256);

  // ---- global head ----
  pool_max<<<gdiv((long)NB * 256), 256, 0, stream>>>(feat2b, pooled, NB, 1024, 256);
  mean_xyz<<<1, 256, 0, stream>>>(xyz2, pxyz, NB, 1024);
  sin_build<<<gdiv((long)16 * 96), 256, 0, stream>>>(pxyz, sinA, 16, 96);
  gemm_wmma<<<1, 256, 0, stream>>>(sinA, Wgpp, gpp.b, gT, 16, 256, 96);
  ln_act_kern<<<1, 256, 0, stream>>>(gT, gpp.g, gpp.be, gBuf, nullptr, NB, 256, 0, 1);
  add_cvt<<<gdiv((long)16 * 256), 256, 0, stream>>>(pooled, gBuf, A2h, 16, 256);
  gemm_wmma<<<1, 256, 0, stream>>>(A2h, Wout, outP.b, outT, 16, 256, 256);
  ln_act_kern<<<1, 256, 0, stream>>>(outT, outP.g, outP.be, (float*)d_out, nullptr, NB, 256, 0, 0);
}
